// Patchwork_85031762526811
// MI455X (gfx1250) — matrix operand order repaired and checked by reading, not compiled, not run on hardware
//
#include <hip/hip_runtime.h>

typedef __bf16 bf16_t;
typedef __attribute__((ext_vector_type(16))) __bf16 v16bf;
typedef __attribute__((ext_vector_type(8)))  __bf16 v8bf;
typedef __attribute__((ext_vector_type(8)))  float  v8f;

#define NB      16384
#define NCOMP   8
#define APC     512
#define DC      512
#define HC      1024
#define MT      64
#define LN_EPSF 1e-5f

// B fragment: 16 contiguous bf16 (32 B) per lane
static __device__ __forceinline__ v16bf load_bfrag(const bf16_t* p) {
  const v8bf* pv = (const v8bf*)p;
  v8bf lo = pv[0], hi = pv[1];
  v16bf r;
#pragma unroll
  for (int i = 0; i < 8; ++i) { r[i] = lo[i]; r[i + 8] = hi[i]; }
  return r;
}

// A fragment: 16 contiguous bf16 (32 B) per lane (K 16h..16h+15)
static __device__ __forceinline__ v16bf load_afrag(const bf16_t* p) {
  const v8bf* pv = (const v8bf*)p;
  v8bf lo = pv[0], hi = pv[1];
  v16bf r;
#pragma unroll
  for (int i = 0; i < 8; ++i) { r[i] = lo[i]; r[i + 8] = hi[i]; }
  return r;
}

// ---------- prologue 1: W1/W2 f32 -> bf16, transposed to [k][n][K] ----------
__global__ void wconv_kernel(const float* __restrict__ W1,
                             const float* __restrict__ W2,
                             bf16_t* __restrict__ w1t,
                             bf16_t* __restrict__ w2t) {
  unsigned tid = blockIdx.x * blockDim.x + threadIdx.x;   // 2^23 threads
  unsigned o   = tid & ((1u << 22) - 1);
  unsigned k   = o >> 19;
  unsigned rem = o & ((1u << 19) - 1);
  if (tid < (1u << 22)) {
    // W1: [k][a(512)][h(1024)] -> w1t: [k][h][a]
    unsigned h = rem >> 9, a = rem & 511u;
    w1t[o] = (bf16_t)W1[((size_t)k << 19) + ((size_t)a << 10) + h];
  } else {
    // W2: [k][h(1024)][d(512)] -> w2t: [k][d][h]
    unsigned d = rem >> 10, h = rem & 1023u;
    w2t[o] = (bf16_t)W2[((size_t)k << 19) + ((size_t)h << 9) + d];
  }
}

// ---------- prologue 2: gather tri columns k::8 -> Xg[k][b][a] (bf16) ----------
__global__ void xgather_kernel(const float* __restrict__ tri,
                               bf16_t* __restrict__ xg) {
  unsigned tid = blockIdx.x * blockDim.x + threadIdx.x;   // 2^23 threads
  unsigned b = tid >> 9, a = tid & 511u;
  const float4* p = (const float4*)(tri + ((size_t)b << 12) + ((size_t)a << 3));
  float4 f0 = p[0], f1 = p[1];
  float v[8] = {f0.x, f0.y, f0.z, f0.w, f1.x, f1.y, f1.z, f1.w};
#pragma unroll
  for (int j = 0; j < 8; ++j)
    xg[((size_t)j << 23) + ((size_t)b << 9) + a] = (bf16_t)v[j];
}

// ---------- fused: GEMM1 + bias + sqrelu + GEMM2 + bias + LayerNorm ----------
__global__ __launch_bounds__(256)
void fused_mlp_kernel(const bf16_t* __restrict__ xg,
                      const bf16_t* __restrict__ w1t,
                      const bf16_t* __restrict__ w2t,
                      const float* __restrict__ b1,
                      const float* __restrict__ b2,
                      const float* __restrict__ gamma,
                      const float* __restrict__ beta,
                      float* __restrict__ out) {
  __shared__ __align__(16) bf16_t sX[MT * APC];            // 64 KB
  __shared__ __align__(16) union SHY {
    bf16_t h[MT * HC];                                     // 128 KB (GEMM1 out)
    float  y[MT * DC];                                     // 128 KB (GEMM2 out)
  } sHY;

  const int k    = blockIdx.y;
  const int row0 = blockIdx.x * MT;
  const int tid  = threadIdx.x;
  const int wav  = tid >> 5;
  const int lane = tid & 31;
  const int lr   = lane & 15;    // row (A) / column (B,C) within 16-tile
  const int ks   = lane >> 4;    // lane-half selector

  // ---- stage X tile [64][512] bf16 into LDS (fully coalesced) ----
  {
    const uint4* src = (const uint4*)(xg + ((size_t)k << 23) + (size_t)row0 * APC);
    uint4* dst = (uint4*)sX;
    for (int i = tid; i < (MT * APC * 2) / 16; i += 256) dst[i] = src[i];
  }
  __syncthreads();

  // ============ GEMM1: h = sqrelu(X @ W1 + b1), bf16 out ============
  // wave owns 128 columns x all 64 rows; each B fragment reused by 4 WMMAs;
  // B fragments double-buffered one K-step ahead.
  {
    const bf16_t* w1k = w1t + ((size_t)k * HC * APC);
    for (int cp = 0; cp < 2; ++cp) {
      const int nbase = (wav * 8 + cp * 4) * 16 + lr;
      v8f acc[4][4] = {};   // [col-tile][row-group]
      v16bf B[2][4];
#pragma unroll
      for (int ct = 0; ct < 4; ++ct)
        B[0][ct] = load_bfrag(&w1k[(size_t)(nbase + ct * 16) * APC + ks * 16]);
#pragma unroll 2
      for (int kt = 0; kt < APC / 32; ++kt) {
        const int cur = kt & 1, nxt = cur ^ 1;
        if (kt + 1 < APC / 32) {
#pragma unroll
          for (int ct = 0; ct < 4; ++ct)
            B[nxt][ct] = load_bfrag(
                &w1k[(size_t)(nbase + ct * 16) * APC + (kt + 1) * 32 + ks * 16]);
        }
        v16bf A[4];
#pragma unroll
        for (int rg = 0; rg < 4; ++rg)
          A[rg] = load_afrag(&sX[(rg * 16 + lr) * APC + kt * 32 + ks * 16]);
#pragma unroll
        for (int ct = 0; ct < 4; ++ct)
#pragma unroll
          for (int rg = 0; rg < 4; ++rg)
            acc[ct][rg] = __builtin_amdgcn_wmma_f32_16x16x32_bf16(
                false, A[rg], false, B[cur][ct], (short)0, acc[ct][rg], false, false);
      }
      // epilogue: + b1, squared ReLU, convert to bf16 into sH
#pragma unroll
      for (int ct = 0; ct < 4; ++ct) {
        int n = nbase + ct * 16;
        float bb = b1[k * HC + n];
#pragma unroll
        for (int rg = 0; rg < 4; ++rg) {
#pragma unroll
          for (int j = 0; j < 8; ++j) {
            int row = rg * 16 + j + 8 * ks;
            float v = acc[ct][rg][j] + bb;
            v = v > 0.f ? v : 0.f;
            sHY.h[row * HC + n] = (bf16_t)(v * v);
          }
        }
      }
    }
  }
  __syncthreads();

  // ============ GEMM2: y = h @ W2 + b2 (f32 kept in regs) ============
  // wave owns 64 columns x all 64 rows in one pass; 4x B reuse; B double-buffered.
  {
    const bf16_t* w2k = w2t + ((size_t)k * DC * HC);
    const int nbase = wav * 64 + lr;
    v8f acc[4][4] = {};   // [col-tile][row-group]
    v16bf B[2][4];
#pragma unroll
    for (int ct = 0; ct < 4; ++ct)
      B[0][ct] = load_bfrag(&w2k[(size_t)(nbase + ct * 16) * HC + ks * 16]);
#pragma unroll 2
    for (int kt = 0; kt < HC / 32; ++kt) {
      const int cur = kt & 1, nxt = cur ^ 1;
      if (kt + 1 < HC / 32) {
#pragma unroll
        for (int ct = 0; ct < 4; ++ct)
          B[nxt][ct] = load_bfrag(
              &w2k[(size_t)(nbase + ct * 16) * HC + (kt + 1) * 32 + ks * 16]);
      }
      v16bf A[4];
#pragma unroll
      for (int rg = 0; rg < 4; ++rg)
        A[rg] = load_afrag(&sHY.h[(rg * 16 + lr) * HC + kt * 32 + ks * 16]);
#pragma unroll
      for (int ct = 0; ct < 4; ++ct)
#pragma unroll
        for (int rg = 0; rg < 4; ++rg)
          acc[ct][rg] = __builtin_amdgcn_wmma_f32_16x16x32_bf16(
              false, A[rg], false, B[cur][ct], (short)0, acc[ct][rg], false, false);
    }
    // add b2 while still in registers
#pragma unroll
    for (int ct = 0; ct < 4; ++ct) {
      float bb = b2[k * DC + nbase + ct * 16];
#pragma unroll
      for (int rg = 0; rg < 4; ++rg)
#pragma unroll
        for (int j = 0; j < 8; ++j) acc[ct][rg][j] += bb;
    }
    __syncthreads();   // everyone done reading sH; safe to overlay f32 y
#pragma unroll
    for (int ct = 0; ct < 4; ++ct) {
      int n = nbase + ct * 16;
#pragma unroll
      for (int rg = 0; rg < 4; ++rg) {
#pragma unroll
        for (int j = 0; j < 8; ++j) {
          int row = rg * 16 + j + 8 * ks;
          sHY.y[row * DC + n] = acc[ct][rg][j];
        }
      }
    }
  }
  __syncthreads();

  // ============ LayerNorm(512) + gamma/beta + store ============
  {
    const float* gk = gamma + k * DC;
    const float* bk = beta + k * DC;
#pragma unroll
    for (int rr = 0; rr < 8; ++rr) {
      int row = wav * 8 + rr;
      float s = 0.f, s2 = 0.f;
#pragma unroll
      for (int i = 0; i < DC / 32; ++i) {
        float v = sHY.y[row * DC + i * 32 + lane];
        s += v; s2 += v * v;
      }
#pragma unroll
      for (int off = 16; off > 0; off >>= 1) {
        s  += __shfl_xor(s,  off, 32);
        s2 += __shfl_xor(s2, off, 32);
      }
      float mu   = s * (1.f / DC);
      float var  = s2 * (1.f / DC) - mu * mu;
      float rstd = rsqrtf(var + LN_EPSF);
      float* orow = out + (size_t)(row0 + row) * (NCOMP * DC) + k * DC;
#pragma unroll
      for (int i = 0; i < DC / 32; ++i) {
        int c = i * 32 + lane;
        float v = sHY.y[row * DC + c];
        orow[c] = (v - mu) * rstd * gk[c] + bk[c];
      }
    }
  }
}

extern "C" void kernel_launch(void* const* d_in, const int* in_sizes, int n_in,
                              void* d_out, int out_size, void* d_ws, size_t ws_size,
                              hipStream_t stream) {
  (void)in_sizes; (void)n_in; (void)out_size; (void)ws_size;
  const float* tri   = (const float*)d_in[0];
  const float* W1    = (const float*)d_in[1];
  const float* b1    = (const float*)d_in[2];
  const float* W2    = (const float*)d_in[3];
  const float* b2    = (const float*)d_in[4];
  const float* gamma = (const float*)d_in[5];
  const float* beta  = (const float*)d_in[6];
  float* out = (float*)d_out;

  char* ws = (char*)d_ws;
  bf16_t* w1t = (bf16_t*)ws;                       // 8 MB  [k][h][a] bf16
  bf16_t* w2t = (bf16_t*)(ws + (8u << 20));        // 8 MB  [k][d][h] bf16
  bf16_t* xg  = (bf16_t*)(ws + (16u << 20));       // 128 MB [k][b][a] bf16

  wconv_kernel<<<(1u << 23) / 256, 256, 0, stream>>>(W1, W2, w1t, w2t);
  xgather_kernel<<<(1u << 23) / 256, 256, 0, stream>>>(tri, xg);

  dim3 grid(NB / MT, NCOMP);
  fused_mlp_kernel<<<grid, 256, 0, stream>>>(xg, w1t, w2t, b1, b2, gamma, beta, out);
}
